// DimeNetPlusPlus_P3M_57904749085210
// MI455X (gfx1250) — matrix operand order repaired and checked by reading, not compiled, not run on hardware
//
#include <hip/hip_runtime.h>
#include <cstdint>

typedef __attribute__((ext_vector_type(16))) _Float16 v16h;
typedef __attribute__((ext_vector_type(8)))  _Float16 v8h;
typedef __attribute__((ext_vector_type(8)))  float    v8f;

#define HCH 256
#define INTC 64

__device__ __forceinline__ float silu_f(float v) {
    return v / (1.0f + __expf(-v));
}

// ---------- small utility kernels ----------

__global__ void k_cvt_f16(const float* __restrict__ src, _Float16* __restrict__ dst, int n) {
    int i = blockIdx.x * blockDim.x + threadIdx.x;
    if (i < n) dst[i] = (_Float16)src[i];
}

__global__ void k_zero(float* __restrict__ p, int n) {
    int i = blockIdx.x * blockDim.x + threadIdx.x;
    if (i < n) p[i] = 0.0f;
}

// W[K,N] f32 (row-major)  ->  Wt[N,K] f16 (row-major over K)
__global__ void k_wt(const float* __restrict__ src, _Float16* __restrict__ dst, int K, int N) {
    int i = blockIdx.x * blockDim.x + threadIdx.x;
    if (i < K * N) {
        int n = i / K, k = i - n * K;
        dst[i] = (_Float16)src[k * N + n];
    }
}

// g = (rbf @ W_rbf1) @ W_rbf2   [E,256] ; one block (256 threads) per edge
__global__ __launch_bounds__(256) void k_rbf_gate(const float* __restrict__ rbf,
                                                  const float* __restrict__ W1,   // [6,8]
                                                  const float* __restrict__ W2,   // [8,256]
                                                  float* __restrict__ g) {
    __shared__ float t8[8];
    int e = blockIdx.x, tid = threadIdx.x;
    if (tid < 8) {
        float a = 0.f;
        #pragma unroll
        for (int r = 0; r < 6; ++r) a += rbf[e * 6 + r] * W1[r * 8 + tid];
        t8[tid] = a;
    }
    __syncthreads();
    float a = 0.f;
    #pragma unroll
    for (int j = 0; j < 8; ++j) a += t8[j] * W2[j * 256 + tid];
    g[(size_t)e * 256 + tid] = a;
}

// ---------- WMMA GEMM: out[E,N] = epilogue( A_f16[E,K] @ Wt_f16[N,K]^T ) ----------
// One wave computes one 16x16 tile; 4 waves per block cover 64 output columns.
template<int K, int N, bool BIAS, bool MULG, bool RES, bool S32, bool S16>
__global__ __launch_bounds__(128) void k_gemm(const _Float16* __restrict__ A,    // [E,K]
                                              const _Float16* __restrict__ Wt,   // [N,K]
                                              const float* __restrict__ bias,    // [N]
                                              const float* __restrict__ gmul,    // [E,N]
                                              const float* __restrict__ res,     // [E,N]
                                              float* __restrict__ out32,
                                              _Float16* __restrict__ out16) {
    const int lane = threadIdx.x & 31;
    const int wave = threadIdx.x >> 5;
    const int m0 = blockIdx.x * 16;
    const int n0 = (blockIdx.y * 4 + wave) * 16;
    const int lo = lane & 15;
    const int hi = lane >> 4;

    const _Float16* Arow = A  + (size_t)(m0 + lo) * K;   // A: lane -> row m, halves split by hi
    const _Float16* Brow = Wt + (size_t)(n0 + lo) * K;   // B: lane -> col n, 16 contig K per hi

    v8f c = {};
    #pragma unroll
    for (int ks = 0; ks < K / 32; ++ks) {
        const int k0 = ks * 32;
        if (ks + 1 < K / 32) {              // speculative prefetch of next K-slab
            __builtin_prefetch(Arow + k0 + 32, 0, 1);
            __builtin_prefetch(Brow + k0 + 32, 0, 1);
        }
        // A-matrix 16x32 f16 layout: elem j -> K = 16*hi + j (contiguous, matches B)
        v16h a = *(const v16h*)(Arow + k0 + hi * 16);
        // B-matrix 32x16 f16 layout: elem j -> K = 16*hi + j, col = lane&15  (contiguous in Wt[N,K])
        v16h b = *(const v16h*)(Brow + k0 + hi * 16);
        c = __builtin_amdgcn_wmma_f32_16x16x32_f16(false, a, false, b, (short)0, c, false, false);
    }

    // C/D layout: VGPR v, lanes 0-15 -> M=v ; lanes 16-31 -> M=8+v ; N = lane&15
    #pragma unroll
    for (int v = 0; v < 8; ++v) {
        const int m = m0 + hi * 8 + v;
        const int n = n0 + lo;
        const size_t o = (size_t)m * N + n;
        float val = c[v];
        if (BIAS) val += bias[n];
        val = silu_f(val);
        if (MULG) val *= gmul[o];
        if (RES)  val += res[o];
        if (S32)  out32[o] = val;
        if (S16)  out16[o] = (_Float16)val;
    }
}

// ---------- triplet message: agg[idx_ji] += ((sbf@W1)@W2) * xdown[idx_kj] ----------
__global__ __launch_bounds__(256) void k_triplet(const float* __restrict__ sbf,     // [T,42]
                                                 const int* __restrict__ idx_kj,
                                                 const int* __restrict__ idx_ji,
                                                 const float* __restrict__ W1,      // [42,8]
                                                 const float* __restrict__ W2,      // [8,64]
                                                 const float* __restrict__ xdown,   // [E,64]
                                                 float* __restrict__ agg,           // [E,64]
                                                 int T_) {
    __shared__ float sW1[42 * 8];
    __shared__ float sW2[8 * 64];
    for (int i = threadIdx.x; i < 42 * 8; i += 256) sW1[i] = W1[i];
    for (int i = threadIdx.x; i < 8 * 64;  i += 256) sW2[i] = W2[i];
    __syncthreads();

    int t = blockIdx.x * blockDim.x + threadIdx.x;
    if (t >= T_) return;

    float t8[8] = {0.f,0.f,0.f,0.f,0.f,0.f,0.f,0.f};
    const float* srow = sbf + (size_t)t * 42;
    for (int r = 0; r < 42; ++r) {
        float s = srow[r];
        #pragma unroll
        for (int j = 0; j < 8; ++j) t8[j] += s * sW1[r * 8 + j];
    }
    const int kj = idx_kj[t], ji = idx_ji[t];
    const float4* xr = (const float4*)(xdown + (size_t)kj * 64);
    float* ar = agg + (size_t)ji * 64;
    #pragma unroll 4
    for (int q = 0; q < 16; ++q) {
        float4 xv = xr[q];
        float m0 = 0.f, m1 = 0.f, m2 = 0.f, m3 = 0.f;
        #pragma unroll
        for (int j = 0; j < 8; ++j) {
            const float* w = sW2 + j * 64 + q * 4;
            const float tj = t8[j];
            m0 += tj * w[0]; m1 += tj * w[1]; m2 += tj * w[2]; m3 += tj * w[3];
        }
        __hip_atomic_fetch_add(ar + q * 4 + 0, m0 * xv.x, __ATOMIC_RELAXED, __HIP_MEMORY_SCOPE_AGENT);
        __hip_atomic_fetch_add(ar + q * 4 + 1, m1 * xv.y, __ATOMIC_RELAXED, __HIP_MEMORY_SCOPE_AGENT);
        __hip_atomic_fetch_add(ar + q * 4 + 2, m2 * xv.z, __ATOMIC_RELAXED, __HIP_MEMORY_SCOPE_AGENT);
        __hip_atomic_fetch_add(ar + q * 4 + 3, m3 * xv.w, __ATOMIC_RELAXED, __HIP_MEMORY_SCOPE_AGENT);
    }
}

// ---------- launcher ----------

extern "C" void kernel_launch(void* const* d_in, const int* in_sizes, int n_in,
                              void* d_out, int out_size, void* d_ws, size_t ws_size,
                              hipStream_t stream) {
    const float* x      = (const float*)d_in[0];
    const float* rbf    = (const float*)d_in[1];
    const float* sbf    = (const float*)d_in[2];
    const int*   idx_kj = (const int*)d_in[3];
    const int*   idx_ji = (const int*)d_in[4];
    const float* W_ji   = (const float*)d_in[5];
    const float* b_ji   = (const float*)d_in[6];
    const float* W_kj   = (const float*)d_in[7];
    const float* b_kj   = (const float*)d_in[8];
    const float* W_rbf1 = (const float*)d_in[9];
    const float* W_rbf2 = (const float*)d_in[10];
    const float* W_sbf1 = (const float*)d_in[11];
    const float* W_sbf2 = (const float*)d_in[12];
    const float* W_down = (const float*)d_in[13];
    const float* W_up   = (const float*)d_in[14];
    const float* Wb1_1  = (const float*)d_in[15];
    const float* bb1_1  = (const float*)d_in[16];
    const float* Wb1_2  = (const float*)d_in[17];
    const float* bb1_2  = (const float*)d_in[18];
    const float* W_lin  = (const float*)d_in[19];
    const float* b_lin  = (const float*)d_in[20];
    const float* Wa1_1  = (const float*)d_in[21];
    const float* ba1_1  = (const float*)d_in[22];
    const float* Wa1_2  = (const float*)d_in[23];
    const float* ba1_2  = (const float*)d_in[24];
    const float* Wa2_1  = (const float*)d_in[25];
    const float* ba2_1  = (const float*)d_in[26];
    const float* Wa2_2  = (const float*)d_in[27];
    const float* ba2_2  = (const float*)d_in[28];

    const int E_ = in_sizes[0] / HCH;   // 131072
    const int T_ = in_sizes[3];         // 1048576
    float* out = (float*)d_out;

    // workspace layout (buffer lifetimes overlapped; ~466 MB total)
    const size_t MB = 1ull << 20;
    char* ws = (char*)d_ws;
    _Float16* x_h   = (_Float16*)(ws + 0);          //  64 MB  f16 x
    float*    g     = (float*)   (ws + 64  * MB);   // 128 MB  rbf gate (dead after x_kj)
    float*    hbuf  = (float*)   (ws + 64  * MB);   //   (reuses g)  running f32 h
    float*    x_ji  = (float*)   (ws + 192 * MB);   // 128 MB  (dead after up-GEMM)
    _Float16* t_h   = (_Float16*)(ws + 192 * MB);   //   (reuses x_ji)  inner residual f16
    _Float16* bufB  = (_Float16*)(ws + 256 * MB);   //   (reuses x_ji upper half)
    _Float16* xg_h  = (_Float16*)(ws + 320 * MB);   //  64 MB  silu(xWkj+b)*g f16 (dead after down)
    _Float16* bufA  = (_Float16*)(ws + 320 * MB);   //   (reuses xg_h)
    float*    xdown = (float*)   (ws + 384 * MB);   //  32 MB  x_kj down-projection
    float*    agg   = (float*)   (ws + 416 * MB);   //  32 MB  segment-sum target
    _Float16* agg_h = (_Float16*)(ws + 448 * MB);   //  16 MB
    _Float16* wt    = (_Float16*)(ws + 464 * MB);   // ~1.2 MB transposed f16 weights

    _Float16* wt_ji   = wt;
    _Float16* wt_kj   = wt + 65536;
    _Float16* wt_down = wt + 131072;
    _Float16* wt_up   = wt + 147456;
    _Float16* wt_b11  = wt + 163840;
    _Float16* wt_b12  = wt + 229376;
    _Float16* wt_lin  = wt + 294912;
    _Float16* wt_a11  = wt + 360448;
    _Float16* wt_a12  = wt + 425984;
    _Float16* wt_a21  = wt + 491520;
    _Float16* wt_a22  = wt + 557056;

    // ---- prep: f16 activations + transposed f16 weights ----
    { int n = E_ * HCH; k_cvt_f16<<<(n + 255) / 256, 256, 0, stream>>>(x, x_h, n); }
    k_wt<<<256, 256, 0, stream>>>(W_ji,   wt_ji,   256, 256);
    k_wt<<<256, 256, 0, stream>>>(W_kj,   wt_kj,   256, 256);
    k_wt<<<64,  256, 0, stream>>>(W_down, wt_down, 256, 64);
    k_wt<<<64,  256, 0, stream>>>(W_up,   wt_up,   64,  256);
    k_wt<<<256, 256, 0, stream>>>(Wb1_1,  wt_b11,  256, 256);
    k_wt<<<256, 256, 0, stream>>>(Wb1_2,  wt_b12,  256, 256);
    k_wt<<<256, 256, 0, stream>>>(W_lin,  wt_lin,  256, 256);
    k_wt<<<256, 256, 0, stream>>>(Wa1_1,  wt_a11,  256, 256);
    k_wt<<<256, 256, 0, stream>>>(Wa1_2,  wt_a12,  256, 256);
    k_wt<<<256, 256, 0, stream>>>(Wa2_1,  wt_a21,  256, 256);
    k_wt<<<256, 256, 0, stream>>>(Wa2_2,  wt_a22,  256, 256);

    const dim3 blk(128);
    const dim3 gridH(E_ / 16, 4);   // N = 256
    const dim3 gridI(E_ / 16, 1);   // N = 64

    // x_ji = silu(x @ W_ji + b_ji)                       -> f32
    k_gemm<256,256,true,false,false,true,false><<<gridH, blk, 0, stream>>>(
        x_h, wt_ji, b_ji, nullptr, nullptr, x_ji, nullptr);
    // g = (rbf @ W_rbf1) @ W_rbf2
    k_rbf_gate<<<E_, 256, 0, stream>>>(rbf, W_rbf1, W_rbf2, g);
    // xg = silu(x @ W_kj + b_kj) * g                     -> f16
    k_gemm<256,256,true,true,false,false,true><<<gridH, blk, 0, stream>>>(
        x_h, wt_kj, b_kj, g, nullptr, nullptr, xg_h);
    // xdown = silu(xg @ W_down)                          -> f32 [E,64]
    k_gemm<256,64,false,false,false,true,false><<<gridI, blk, 0, stream>>>(
        xg_h, wt_down, nullptr, nullptr, nullptr, xdown, nullptr);

    // agg = segment_sum( ((sbf@W_sbf1)@W_sbf2) * xdown[idx_kj], idx_ji )
    { int n = E_ * INTC; k_zero<<<(n + 255) / 256, 256, 0, stream>>>(agg, n); }
    k_triplet<<<(T_ + 255) / 256, 256, 0, stream>>>(
        sbf, idx_kj, idx_ji, W_sbf1, W_sbf2, xdown, agg, T_);
    { int n = E_ * INTC; k_cvt_f16<<<(n + 255) / 256, 256, 0, stream>>>(agg, agg_h, n); }

    // h = x_ji + silu(agg @ W_up)                        -> hbuf f32, bufA f16
    k_gemm<64,256,false,false,true,true,true><<<gridH, blk, 0, stream>>>(
        agg_h, wt_up, nullptr, nullptr, x_ji, hbuf, bufA);
    // t = silu(h @ Wb1_1 + b)                            -> t_h f16
    k_gemm<256,256,true,false,false,false,true><<<gridH, blk, 0, stream>>>(
        bufA, wt_b11, bb1_1, nullptr, nullptr, nullptr, t_h);
    // h = h + silu(t @ Wb1_2 + b)                        -> hbuf f32, bufA f16
    k_gemm<256,256,true,false,true,true,true><<<gridH, blk, 0, stream>>>(
        t_h, wt_b12, bb1_2, nullptr, hbuf, hbuf, bufA);
    // h = silu(h @ W_lin + b) + x                        -> hbuf f32, bufB f16
    k_gemm<256,256,true,false,true,true,true><<<gridH, blk, 0, stream>>>(
        bufA, wt_lin, b_lin, nullptr, x, hbuf, bufB);
    // t = silu(h @ Wa1_1 + b)
    k_gemm<256,256,true,false,false,false,true><<<gridH, blk, 0, stream>>>(
        bufB, wt_a11, ba1_1, nullptr, nullptr, nullptr, t_h);
    // h = h + silu(t @ Wa1_2 + b)
    k_gemm<256,256,true,false,true,true,true><<<gridH, blk, 0, stream>>>(
        t_h, wt_a12, ba1_2, nullptr, hbuf, hbuf, bufB);
    // t = silu(h @ Wa2_1 + b)
    k_gemm<256,256,true,false,false,false,true><<<gridH, blk, 0, stream>>>(
        bufB, wt_a21, ba2_1, nullptr, nullptr, nullptr, t_h);
    // out = h + silu(t @ Wa2_2 + b)                      -> d_out f32
    k_gemm<256,256,true,false,true,true,false><<<gridH, blk, 0, stream>>>(
        t_h, wt_a22, ba2_2, nullptr, hbuf, out, nullptr);

    (void)n_in; (void)out_size; (void)ws_size;
}
